// FlashAttentionV2_58841051955440
// MI455X (gfx1250) — matrix operand order repaired and checked by reading, not compiled, not run on hardware
//
#include <hip/hip_runtime.h>
#include <hip/hip_bf16.h>
#include <stdint.h>

#define TSEQ   2048
#define DMODEL 1024
#define NHEADS 16
#define DHEAD  64
#define BATCH  2
#define NTOK   (BATCH * TSEQ)   // 4096

typedef __attribute__((ext_vector_type(16))) __bf16 v16bf;
typedef __attribute__((ext_vector_type(8)))  __bf16 v8bf;
typedef __attribute__((ext_vector_type(8)))  float  v8f;

union AFrag { v16bf v; v8bf h[2]; };

__device__ __forceinline__ uint16_t f2bf(float f) {
  uint32_t x = __builtin_bit_cast(uint32_t, f);
  uint32_t r = (x + 0x7FFFu + ((x >> 16) & 1u)) >> 16;   // round-to-nearest-even
  return (uint16_t)r;
}

__device__ __forceinline__ float xor16(float v) {
  int i = __builtin_bit_cast(int, v);
  i = __builtin_amdgcn_ds_swizzle(i, 0x401f);            // group-of-32 SWAPX16
  return __builtin_bit_cast(float, i);
}

__device__ __forceinline__ v8f wmma_bf16(v16bf a, v16bf b, v8f c) {
  return __builtin_amdgcn_wmma_f32_16x16x32_bf16(false, a, false, b, (short)0, c,
                                                 false, false);
}

// Async 16B/lane copy global -> LDS (VDST = per-lane LDS byte offset). Tracked
// by ASYNCcnt; completion is in-order, so s_wait_asynccnt N retires older ops.
__device__ __forceinline__ void async_cp16(uint32_t lds_off, const void* gptr) {
  asm volatile("global_load_async_to_lds_b128 %0, %1, off"
               :: "v"(lds_off), "v"((uint64_t)(uintptr_t)gptr) : "memory");
}
#define WAIT_ASYNCCNT(n) asm volatile("s_wait_asynccnt " #n ::: "memory")
__device__ __forceinline__ uint32_t lds_off_of(const void* p) {
  return (uint32_t)(uintptr_t)p;   // low 32 bits of LDS-aperture flat addr
}

// ---------------------------------------------------------------------------
// fp32 -> bf16 convert, optionally transposing (weights: Wt[k][n] = W[n][k])
// ---------------------------------------------------------------------------
__global__ __launch_bounds__(256) void convert_cast_kernel(
    const float* __restrict__ src, uint16_t* __restrict__ dst,
    int rows, int cols, int do_transpose) {
  int i = blockIdx.x * 256 + threadIdx.x;
  if (i >= rows * cols) return;
  int r = i / cols, c = i % cols;
  uint16_t b = f2bf(src[i]);
  dst[do_transpose ? (c * rows + r) : i] = b;
}

// ---------------------------------------------------------------------------
// GEMM: C[m,n] = sum_k A[m,k] * Bt[k,n]. Workgroup tile 128(m) x 64(n), 8 waves
// each own 16x64. B (32k x 64n = 4KB) is double-buffered in LDS via async copy
// and shared by all 8 waves; A streams from global/L2 (unique rows per wave).
// mode 0: bf16 [b,h,t,dh]  | mode 1: bf16 [b,h,dh,t]*scale | mode 2: fp32 [m,n]
// ---------------------------------------------------------------------------
__global__ __launch_bounds__(256) void gemm_bf16_kernel(
    const uint16_t* __restrict__ A, const uint16_t* __restrict__ Bt,
    void* __restrict__ Out, int mode, float scale) {
  __shared__ __align__(32) uint16_t ldsB[2][32 * 64];

  const int lane = threadIdx.x & 31;
  const int wv   = threadIdx.x >> 5;
  const int m0   = (blockIdx.x >> 4) * 128 + wv * 16;   // 32 m-blocks
  const int n0   = (blockIdx.x & 15) * 64;              // 16 n-blocks
  const int lm   = lane & 15;
  const int hi   = lane >> 4;
  const int kg0  = hi ? 8 : 0;

  // cooperative async slice: thread t moves 16B of the 32x64 B tile
  const int brow = threadIdx.x >> 3;          // 0..31 (k row)
  const int bcol = (threadIdx.x & 7) * 8;     // 0..56 (n col)
  const uint16_t* bsrc = Bt + (size_t)brow * DMODEL + n0 + bcol;
  const uint32_t boff[2] = { lds_off_of(&ldsB[0][brow * 64 + bcol]),
                             lds_off_of(&ldsB[1][brow * 64 + bcol]) };

  async_cp16(boff[0], bsrc);                              // k-block 0
  async_cp16(boff[1], bsrc + 32 * DMODEL);                // k-block 1

  v8f acc[4] = {v8f{}, v8f{}, v8f{}, v8f{}};
  const uint16_t* arow = A + (size_t)(m0 + lm) * DMODEL;

  for (int kb = 0; kb < 32; ++kb) {
    const int cur = kb & 1;
    if (kb < 31) { WAIT_ASYNCCNT(1); } else { WAIT_ASYNCCNT(0); }
    __syncthreads();                                      // tile `cur` visible

    const int k0 = kb * 32;
    AFrag a;
    a.h[0] = *(const v8bf*)(arow + k0 + kg0);
    a.h[1] = *(const v8bf*)(arow + k0 + kg0 + 16);
    const uint16_t* bl = &ldsB[cur][lm];                  // lane = n column
#pragma unroll
    for (int g = 0; g < 4; ++g) {
      v16bf b;
#pragma unroll
      for (int s = 0; s < 16; ++s)
        b[s] = __builtin_bit_cast(__bf16, bl[((s & 7) + kg0 + ((s >> 3) << 4)) * 64 + g * 16]);
      acc[g] = wmma_bf16(a.v, b, acc[g]);
    }

    __syncthreads();                                      // everyone done w/ cur
    if (kb + 2 < 32)
      async_cp16(boff[cur], bsrc + (size_t)(kb + 2) * 32 * DMODEL);
  }

#pragma unroll
  for (int g = 0; g < 4; ++g) {
    int n_g = n0 + g * 16 + lm;
#pragma unroll
    for (int r = 0; r < 8; ++r) {
      int m_g = m0 + r + 8 * hi;
      float val = acc[g][r] * scale;
      if (mode == 2) {
        ((float*)Out)[(size_t)m_g * DMODEL + n_g] = val;
      } else {
        int b = m_g >> 11, t = m_g & (TSEQ - 1);
        int h = n_g >> 6,  d = n_g & (DHEAD - 1);
        uint16_t bv = f2bf(val);
        if (mode == 0)
          ((uint16_t*)Out)[((size_t)(b * NHEADS + h) * TSEQ + t) * DHEAD + d] = bv;
        else
          ((uint16_t*)Out)[((size_t)(b * NHEADS + h) * DHEAD + d) * TSEQ + t] = bv;
      }
    }
  }
}

// ---------------------------------------------------------------------------
// Flash attention. Workgroup = 128 contiguous queries of one (b,h); 8 waves
// each own 16 queries. K (32x64) and V^T (64x32) key-tiles are double-buffered
// in LDS via async copy and shared by all waves. Each wave computes
// S^T = K*Q^T (per-lane-scalar online softmax), reshapes P^T through its LDS
// slot, then out^T += V^T * P^T. Causality: low-q waves skip compute but keep
// the uniform barrier/staging schedule.
// Qt,Vt: bf16 [b,h,dh,T]; Kn: bf16 [b,h,t,dh]; AttOut: bf16 [b,t,h*64+dh].
// ---------------------------------------------------------------------------
__global__ __launch_bounds__(256) void flash_attn_kernel(
    const uint16_t* __restrict__ Qt, const uint16_t* __restrict__ Kn,
    const uint16_t* __restrict__ Vt, uint16_t* __restrict__ AttOut) {
  __shared__ __align__(32) uint16_t ldsK[2][32 * 64];   // [key][dh]
  __shared__ __align__(32) uint16_t ldsV[2][64 * 32];   // [dh][key]
  __shared__ __align__(32) uint16_t ldsP[8][32 * 16];   // per-wave P^T

  const int lane = threadIdx.x & 31;
  const int wv   = threadIdx.x >> 5;
  const int qblk = blockIdx.x & 15;                 // T/128 = 16
  const int h    = (blockIdx.x >> 4) & (NHEADS - 1);
  const int b    = blockIdx.x >> 8;
  const int q0   = (qblk * 8 + wv) * 16;
  const int lm   = lane & 15;
  const int hi   = lane >> 4;
  const int kg0  = hi ? 8 : 0;
  const float NEG_INF = -__builtin_inff();

  const uint16_t* qtp = Qt + (size_t)((b * NHEADS + h) * DHEAD) * TSEQ;
  const uint16_t* knp = Kn + (size_t)((b * NHEADS + h) * TSEQ) * DHEAD;
  const uint16_t* vtp = Vt + (size_t)((b * NHEADS + h) * DHEAD) * TSEQ;

  // cooperative async slices (16B per thread per tile)
  const int krow = threadIdx.x >> 3, kcol = (threadIdx.x & 7) * 8;  // K tile
  const int vrow = threadIdx.x >> 2, vcol = (threadIdx.x & 3) * 8;  // V^T tile
  const uint16_t* ksrc = knp + (size_t)krow * DHEAD + kcol;
  const uint16_t* vsrc = vtp + (size_t)vrow * TSEQ + vcol;
  const uint32_t koff[2] = { lds_off_of(&ldsK[0][krow * 64 + kcol]),
                             lds_off_of(&ldsK[1][krow * 64 + kcol]) };
  const uint32_t voff[2] = { lds_off_of(&ldsV[0][vrow * 32 + vcol]),
                             lds_off_of(&ldsV[1][vrow * 32 + vcol]) };

  // Q^T B-operands (lane = query column, dh in table order). Loaded once.
  v16bf bq0, bq1;
#pragma unroll
  for (int s = 0; s < 16; ++s) {
    int kd = (s & 7) + kg0 + ((s >> 3) << 4);
    bq0[s] = __builtin_bit_cast(__bf16, qtp[(size_t)kd * TSEQ + q0 + lm]);
    bq1[s] = __builtin_bit_cast(__bf16, qtp[(size_t)(kd + 32) * TSEQ + q0 + lm]);
  }

  v8f o[4] = {v8f{}, v8f{}, v8f{}, v8f{}};
  float mrun = NEG_INF, lrun = 0.f;
  const int qL = q0 + lm;
  uint16_t* pbase = &ldsP[wv][0];

  const int nkb   = 4 * (qblk + 1);            // block-level key blocks of 32
  const int nkb_w = (q0 + 16 + 31) >> 5;       // this wave's active blocks

  async_cp16(koff[0], ksrc);  async_cp16(voff[0], vsrc);
  if (nkb > 1) { async_cp16(koff[1], ksrc + (size_t)32 * DHEAD);
                 async_cp16(voff[1], vsrc + 32); }

  for (int kb = 0; kb < nkb; ++kb) {
    const int cur = kb & 1;
    const int kk0 = kb * 32;
    if (kb + 1 < nkb) { WAIT_ASYNCCNT(2); } else { WAIT_ASYNCCNT(0); }
    __syncthreads();                            // K/V tile `cur` visible

    if (kb < nkb_w) {
      // ---- S^T = K (A, from LDS) x Q^T (B)
      v8f st0 = {}, st1 = {};
      const uint16_t* kr0 = &ldsK[cur][(size_t)lm * 64];
      const uint16_t* kr1 = &ldsK[cur][(size_t)(16 + lm) * 64];
      AFrag a;
      a.h[0] = *(const v8bf*)(kr0 + kg0);      a.h[1] = *(const v8bf*)(kr0 + kg0 + 16);
      st0 = wmma_bf16(a.v, bq0, st0);
      a.h[0] = *(const v8bf*)(kr0 + 32 + kg0); a.h[1] = *(const v8bf*)(kr0 + 32 + kg0 + 16);
      st0 = wmma_bf16(a.v, bq1, st0);
      a.h[0] = *(const v8bf*)(kr1 + kg0);      a.h[1] = *(const v8bf*)(kr1 + kg0 + 16);
      st1 = wmma_bf16(a.v, bq0, st1);
      a.h[0] = *(const v8bf*)(kr1 + 32 + kg0); a.h[1] = *(const v8bf*)(kr1 + 32 + kg0 + 16);
      st1 = wmma_bf16(a.v, bq1, st1);

      // ---- causal mask + per-query online softmax (lane owns query qL)
      float lmax = NEG_INF;
#pragma unroll
      for (int r = 0; r < 8; ++r) {
        int key0 = kk0 + r + 8 * hi;
        float s0 = (key0 <= qL) ? st0[r] : NEG_INF;
        float s1 = (key0 + 16 <= qL) ? st1[r] : NEG_INF;
        st0[r] = s0; st1[r] = s1;
        lmax = fmaxf(lmax, fmaxf(s0, s1));
      }
      lmax = fmaxf(lmax, xor16(lmax));
      float mnew  = fmaxf(mrun, lmax);
      float alpha = __expf(mrun - mnew);
      float lsum  = 0.f;
#pragma unroll
      for (int r = 0; r < 8; ++r) {
        float e0 = __expf(st0[r] - mnew);
        float e1 = __expf(st1[r] - mnew);
        lsum += e0 + e1;
        pbase[lm * 32 + r + 8 * hi]      = f2bf(e0);   // P[q][key_local]
        pbase[lm * 32 + 16 + r + 8 * hi] = f2bf(e1);
      }
      lsum += xor16(lsum);
      lrun = lrun * alpha + lsum;
      mrun = mnew;
#pragma unroll
      for (int g = 0; g < 4; ++g)
#pragma unroll
        for (int r = 0; r < 8; ++r) o[g][r] *= alpha;

      // wave-private LDS: P^T stores must land before B-operand re-read
      asm volatile("s_wait_dscnt 0" ::: "memory");
      AFrag bpf;                                         // lane = query column
      bpf.h[0] = *(const v8bf*)(pbase + lm * 32 + kg0);
      bpf.h[1] = *(const v8bf*)(pbase + lm * 32 + kg0 + 16);
      v16bf bp = bpf.v;

      // ---- out^T += V^T (A, from LDS) x P^T (B)
#pragma unroll
      for (int g = 0; g < 4; ++g) {
        const uint16_t* vr = &ldsV[cur][(size_t)(g * 16 + lm) * 32];
        AFrag av;
        av.h[0] = *(const v8bf*)(vr + kg0);
        av.h[1] = *(const v8bf*)(vr + kg0 + 16);
        o[g] = wmma_bf16(av.v, bp, o[g]);
      }
    }

    __syncthreads();                            // all waves done with `cur`
    if (kb + 2 < nkb) {
      async_cp16(koff[cur], ksrc + (size_t)(kk0 + 64) * DHEAD);
      async_cp16(voff[cur], vsrc + (kk0 + 64));
    }
  }

  // ---- finalize: /(l+eps), store concat-head bf16 [b, t, h*64+dh]
  float inv = 1.f / (lrun + 1e-9f);
  uint16_t* orow = AttOut + (size_t)(b * TSEQ + qL) * DMODEL + h * DHEAD;
#pragma unroll
  for (int g = 0; g < 4; ++g)
#pragma unroll
    for (int r = 0; r < 8; ++r)
      orow[g * 16 + r + 8 * hi] = f2bf(o[g][r] * inv);
}

// ---------------------------------------------------------------------------
extern "C" void kernel_launch(void* const* d_in, const int* in_sizes, int n_in,
                              void* d_out, int out_size, void* d_ws, size_t ws_size,
                              hipStream_t stream) {
  const float* x  = (const float*)d_in[0];
  const float* Wq = (const float*)d_in[1];
  const float* Wk = (const float*)d_in[2];
  const float* Wv = (const float*)d_in[3];
  const float* Wo = (const float*)d_in[4];

  uint8_t* ws = (uint8_t*)d_ws;
  uint16_t* xb   = (uint16_t*)(ws);                        //  8 MB  x  bf16
  uint16_t* wtq  = (uint16_t*)(ws + ( 8ull << 20));        //  2 MB  Wq^T
  uint16_t* wtk  = (uint16_t*)(ws + (10ull << 20));        //  2 MB  Wk^T
  uint16_t* wtv  = (uint16_t*)(ws + (12ull << 20));        //  2 MB  Wv^T
  uint16_t* wto  = (uint16_t*)(ws + (14ull << 20));        //  2 MB  Wo^T
  uint16_t* Kmat = (uint16_t*)(ws + (16ull << 20));        //  8 MB  K  [b,h,t,dh]
  uint16_t* Qt   = (uint16_t*)(ws + (24ull << 20));        //  8 MB  Q^T[b,h,dh,t]
  uint16_t* Vt   = (uint16_t*)(ws + (32ull << 20));        //  8 MB  V^T[b,h,dh,t]
  uint16_t* att  = (uint16_t*)(ws + (40ull << 20));        //  8 MB  attn out

  convert_cast_kernel<<<(NTOK * DMODEL + 255) / 256, 256, 0, stream>>>(x, xb, NTOK, DMODEL, 0);
  convert_cast_kernel<<<(DMODEL * DMODEL + 255) / 256, 256, 0, stream>>>(Wq, wtq, DMODEL, DMODEL, 1);
  convert_cast_kernel<<<(DMODEL * DMODEL + 255) / 256, 256, 0, stream>>>(Wk, wtk, DMODEL, DMODEL, 1);
  convert_cast_kernel<<<(DMODEL * DMODEL + 255) / 256, 256, 0, stream>>>(Wv, wtv, DMODEL, DMODEL, 1);
  convert_cast_kernel<<<(DMODEL * DMODEL + 255) / 256, 256, 0, stream>>>(Wo, wto, DMODEL, DMODEL, 1);

  gemm_bf16_kernel<<<512, 256, 0, stream>>>(xb, wtq, Qt,   1, 0.125f);  // Q/sqrt(dh)
  gemm_bf16_kernel<<<512, 256, 0, stream>>>(xb, wtk, Kmat, 0, 1.0f);
  gemm_bf16_kernel<<<512, 256, 0, stream>>>(xb, wtv, Vt,   1, 1.0f);

  flash_attn_kernel<<<512, 256, 0, stream>>>(Qt, Kmat, Vt, att);

  gemm_bf16_kernel<<<512, 256, 0, stream>>>(att, wto, d_out, 2, 1.0f);
}
